// RNNContext_2602750181977
// MI455X (gfx1250) — matrix operand order repaired and checked by reading, not compiled, not run on hardware
//
#include <hip/hip_runtime.h>

#define DI 64
#define DH 128
#define DT 200
#define DB 4096

typedef __attribute__((ext_vector_type(16))) __bf16        v16bf;
typedef __attribute__((ext_vector_type(8)))  float         v8f;
typedef __attribute__((ext_vector_type(4)))  float         f32x4;
typedef __attribute__((ext_vector_type(4)))  unsigned int  u32x4;

// workspace layout: packed bf16 B-fragments + fused biases
#define WIH_OFF  0                         // 24 n-tiles * 2 k-chunks * 1KB
#define WHH_OFF  (24 * 2 * 1024)           // 24 n-tiles * 4 k-chunks * 1KB
#define BIAS_OFF (WHH_OFF + 24 * 4 * 1024) // brz[256], bxn[128], bhn[128] f32

union VB {
  v16bf v;
  __bf16 e[16];
  u32x4 q[2];
};

#define WMMA_BF16(A, B, C) \
  __builtin_amdgcn_wmma_f32_16x16x32_bf16(false, (A), false, (B), (short)0, (C), false, false)

// WGP-scope prefetch (scope field 0): fills all cache levels incl. WGP$.
// (__builtin_prefetch maps to SCOPE_SE, which skips the WGP cache.)
#define PREFETCH_WGP(p) asm volatile("global_prefetch_b8 %0, off" ::"v"(p))

// scheduling pipeline directives
#define SGB(mask, size) __builtin_amdgcn_sched_group_barrier((mask), (size), 0)

__device__ __forceinline__ v8f splat8(float s) {
  v8f v;
#pragma unroll
  for (int i = 0; i < 8; ++i) v[i] = s;
  return v;
}

__device__ __forceinline__ float fast_sigmoid(float x) {
  return __builtin_amdgcn_rcpf(1.0f + __expf(-x));
}

__device__ __forceinline__ float fast_tanh(float x) {
  float e = __expf(-2.0f * x);
  return (1.0f - e) * __builtin_amdgcn_rcpf(1.0f + e);
}

// ---------------------------------------------------------------------------
// One-shot pack: weights -> bf16 B-fragment layout in workspace (L2 resident).
// B-fragment (16x16x32 bf16 wmma): lane L holds N = L%16, K = (L<16?0:16)+i,
// 16 bf16 = 32 contiguous bytes per lane, 1KB per fragment block.
// ---------------------------------------------------------------------------
__global__ __launch_bounds__(32) void gru_pack_kernel(
    const float* __restrict__ W_ih, const float* __restrict__ W_hh,
    const float* __restrict__ b_ih, const float* __restrict__ b_hh,
    unsigned char* __restrict__ ws) {
  const int bid = blockIdx.x;
  const int lane = threadIdx.x;
  const int ln = lane & 15;
  const int kh = (lane >> 4) ? 16 : 0;

  if (bid < 48) {  // W_ih fragments: block = nt*2 + kc
    const int nt = bid >> 1, kc = bid & 1;
    const int n = nt * 16 + ln;
    const int kb = kc * 32 + kh;
    VB u;
#pragma unroll
    for (int i = 0; i < 16; ++i) u.e[i] = (__bf16)W_ih[n * DI + kb + i];
    *(v16bf*)(ws + WIH_OFF + bid * 1024 + lane * 32) = u.v;
  } else if (bid < 144) {  // W_hh fragments: block = nt*4 + kc
    const int hb = bid - 48;
    const int nt = hb >> 2, kc = hb & 3;
    const int n = nt * 16 + ln;
    const int kb = kc * 32 + kh;
    VB u;
#pragma unroll
    for (int i = 0; i < 16; ++i) u.e[i] = (__bf16)W_hh[n * DH + kb + i];
    *(v16bf*)(ws + WHH_OFF + hb * 1024 + lane * 32) = u.v;
  } else {  // biases: r/z fused (b_ih + b_hh), n kept split
    float* bias = (float*)(ws + BIAS_OFF);
    for (int i = lane; i < 256; i += 32) bias[i] = b_ih[i] + b_hh[i];
    for (int i = lane; i < 128; i += 32) bias[256 + i] = b_ih[256 + i];
    for (int i = lane; i < 128; i += 32) bias[384 + i] = b_hh[256 + i];
  }
}

// ---------------------------------------------------------------------------
// Persistent GRU: block = 4 waves, each wave owns 16 batch rows (one M-tile),
// keeps h in registers (D layout) + mirrored in per-wave LDS (A layout).
// ---------------------------------------------------------------------------
__global__ __launch_bounds__(128, 1) void gru_kernel(
    const float* __restrict__ x, const int* __restrict__ seqlen,
    const unsigned char* __restrict__ ws, float* __restrict__ out) {
  __shared__ __align__(16) __bf16 hbuf[4][16][DH + 8];  // +8 pad: bank stride 68 dw

  const int wave = threadIdx.x >> 5;
  const int lane = threadIdx.x & 31;
  const int lh = lane >> 4;     // half-wave select
  const int ln = lane & 15;     // column / row within tile
  const int b0 = (blockIdx.x * 4 + wave) * 16;

  const float* bias = (const float*)(ws + BIAS_OFF);

  // zero this wave's LDS h region (h0 = 0)
  {
    u32x4* p = (u32x4*)&hbuf[wave][0][0];
    u32x4 z = {0u, 0u, 0u, 0u};
    for (int i = lane; i < 16 * (DH + 8) * 2 / 16; i += 32) p[i] = z;
  }

  // sequence lengths for the 8 rows this lane contributes to (D layout rows)
  int sl[8];
#pragma unroll
  for (int j = 0; j < 8; ++j) sl[j] = seqlen[b0 + lh * 8 + j];

  // per-column bias values (broadcast across the 8 accumulator rows)
  float br[8], bz[8], bxn[8], bhn[8];
#pragma unroll
  for (int c = 0; c < 8; ++c) {
    br[c] = bias[c * 16 + ln];
    bz[c] = bias[128 + c * 16 + ln];
    bxn[c] = bias[256 + c * 16 + ln];
    bhn[c] = bias[384 + c * 16 + ln];
  }

  // rows with seq_len == 0 keep hn = 0: write them now (d_out is poisoned)
#pragma unroll
  for (int j = 0; j < 8; ++j)
    if (sl[j] == 0) {
      const int b = b0 + lh * 8 + j;
#pragma unroll
      for (int c = 0; c < 8; ++c) out[b * DH + c * 16 + ln] = 0.0f;
    }

  // h state in registers, D layout: h[c][j] = row (lh*8+j), col (c*16+ln)
  v8f h[8];
#pragma unroll
  for (int c = 0; c < 8; ++c) h[c] = splat8(0.0f);

  // Opaque per-iteration offset (stays 0): blocks LICM from hoisting the
  // 147KB of weight loads out of the t-loop, while keeping the pointer chain
  // rooted at the kernarg so addrspace(1) inference yields global_load.
  unsigned woff = 0;

#pragma unroll 1
  for (int t = 0; t < DT; ++t) {
    asm volatile("" : "+s"(woff));  // compiler must assume it changes each t
    const unsigned char* wihv = ws + WIH_OFF + woff;
    const unsigned char* whhv = ws + WHH_OFF + woff;

    // ---- prefetch next timestep's x rows into WGP$ (no VGPR cost) ----
    if (t + 1 < DT) {
      const float* xn = x + ((b0 + ln) * DT + (t + 1)) * DI;
      PREFETCH_WGP(xn + lh * 32);  // 32 lanes cover 16 rows x 256B
    }

    // ---- A fragments: x_t (global f32 -> bf16) and h (LDS bf16) ----
    v16bf ax[2], ah[4];
    {
      const float* xp = x + ((b0 + ln) * DT + t) * DI;
#pragma unroll
      for (int kc = 0; kc < 2; ++kc) {
        const float* pa = xp + kc * 32 + lh * 16;       // K group A
        const float* pb = xp + kc * 32 + lh * 16 + 8;  // K group B
        f32x4 a0 = *(const f32x4*)(pa);
        f32x4 a1 = *(const f32x4*)(pa + 4);
        f32x4 g0 = *(const f32x4*)(pb);
        f32x4 g1 = *(const f32x4*)(pb + 4);
        VB u;
#pragma unroll
        for (int i = 0; i < 4; ++i) {
          u.e[i] = (__bf16)a0[i];
          u.e[4 + i] = (__bf16)a1[i];
          u.e[8 + i] = (__bf16)g0[i];
          u.e[12 + i] = (__bf16)g1[i];
        }
        ax[kc] = u.v;
      }
      const __bf16* hr = &hbuf[wave][ln][0];
#pragma unroll
      for (int kc = 0; kc < 4; ++kc) {
        VB u;
        u.q[0] = *(const u32x4*)(hr + kc * 32 + lh * 16);
        u.q[1] = *(const u32x4*)(hr + kc * 32 + lh * 16 + 8);
        ah[kc] = u.v;
      }
    }
    __builtin_amdgcn_sched_barrier(0);

    // ---- 8 output-column chunks over H; gates r(nt=c), z(nt=8+c), n(nt=16+c)
#pragma unroll
    for (int c = 0; c < 8; ++c) {
      // prime next chunk's weight tiles into WGP$: lane-spread addresses
      // cover a whole tile per prefetch instruction (2KB ih / 4KB hh)
      {
        const int cn = (c + 1) & 7;
#pragma unroll
        for (int g = 0; g < 3; ++g) {
          PREFETCH_WGP(wihv + (g * 8 + cn) * 2048 + lane * 64);
          PREFETCH_WGP(whhv + (g * 8 + cn) * 4096 + lane * 128);
        }
      }

      v8f ar = splat8(br[c]);
      v8f az = splat8(bz[c]);
      v8f axv = splat8(bxn[c]);
      v8f ahv = splat8(bhn[c]);

      // 18 fragments, kc-major / gate-minor:
      //   i in [0,6):  ih, kc = i/3, gate = i%3
      //   i in [6,18): hh, kc = (i-6)/3, gate = (i-6)%3
      auto fragAddr = [&](int i) -> const v16bf* {
        return (i < 6)
          ? (const v16bf*)(wihv + (((i % 3) * 8 + c) * 2 + (i / 3)) * 1024 + lane * 32)
          : (const v16bf*)(whhv + ((((i - 6) % 3) * 8 + c) * 4 + ((i - 6) / 3)) * 1024 + lane * 32);
      };

      // software-pipelined fragment ring: 4 fragments in flight
      v16bf Bq[4];
#pragma unroll
      for (int i = 0; i < 4; ++i) Bq[i] = *fragAddr(i);
#pragma unroll
      for (int i = 0; i < 18; ++i) {
        const int g = (i < 6) ? (i % 3) : ((i - 6) % 3);
        const v16bf A = (i < 6) ? ax[i / 3] : ah[(i - 6) / 3];
        if (g == 0)      ar  = WMMA_BF16(A, Bq[i & 3], ar);
        else if (g == 1) az  = WMMA_BF16(A, Bq[i & 3], az);
        else if (i < 6)  axv = WMMA_BF16(A, Bq[i & 3], axv);
        else             ahv = WMMA_BF16(A, Bq[i & 3], ahv);
        if (i + 4 < 18) Bq[i & 3] = *fragAddr(i + 4);
      }

      // gates + state update; mirror new h into LDS (A-source for t+1)
      v8f hn;
#pragma unroll
      for (int j = 0; j < 8; ++j) {
        float r = fast_sigmoid(ar[j]);
        float z = fast_sigmoid(az[j]);
        float n = fast_tanh(axv[j] + r * ahv[j]);
        float hv = n + z * (h[c][j] - n);  // (1-z)*n + z*h
        hn[j] = hv;
        hbuf[wave][lh * 8 + j][c * 16 + ln] = (__bf16)hv;
      }
      h[c] = hn;

      // Pin the region's schedule to a VMEM/WMMA pipeline:
      //   8 loads (4 frags) ahead, then 14 x [1 wmma, 2 loads], then 4 wmma.
      // (region has exactly 36 VMEM-reads and 18 WMMAs; VALU/DS unconstrained)
      SGB(0x20, 8);
#pragma unroll
      for (int i = 0; i < 14; ++i) {
        SGB(0x8, 1);
        SGB(0x20, 2);
      }
      SGB(0x8, 4);

      __builtin_amdgcn_sched_barrier(0);  // region boundary at chunk scope
    }

    // ---- hn capture: row b finalizes exactly when t+1 == seq_len[b] ----
    int hit = 0;
#pragma unroll
    for (int j = 0; j < 8; ++j) hit |= (t + 1 == sl[j]);
    if (__any(hit)) {
#pragma unroll
      for (int j = 0; j < 8; ++j)
        if (t + 1 == sl[j]) {
          const int b = b0 + lh * 8 + j;
#pragma unroll
          for (int c = 0; c < 8; ++c) out[b * DH + c * 16 + ln] = h[c][j];
        }
    }
  }
}

extern "C" void kernel_launch(void* const* d_in, const int* in_sizes, int n_in,
                              void* d_out, int out_size, void* d_ws, size_t ws_size,
                              hipStream_t stream) {
  const float* x = (const float*)d_in[0];
  const int* sl = (const int*)d_in[1];
  const float* W_ih = (const float*)d_in[2];
  const float* W_hh = (const float*)d_in[3];
  const float* b_ih = (const float*)d_in[4];
  const float* b_hh = (const float*)d_in[5];
  unsigned char* ws = (unsigned char*)d_ws;

  gru_pack_kernel<<<145, 32, 0, stream>>>(W_ih, W_hh, b_ih, b_hh, ws);
  gru_kernel<<<DB / 64, 128, 0, stream>>>(x, sl, ws, (float*)d_out);
}
